// CMFDNet_81372450390308
// MI455X (gfx1250) — matrix operand order repaired and checked by reading, not compiled, not run on hardware
//
#include <hip/hip_runtime.h>

// ---------------------------------------------------------------------------
// CMFD-Net pipeline for MI455X (gfx1250, wave32, WMMA)
//   1) normalize features over C, convert to f16 (padded to 1376 rows)
//   2) self-correlation Gram matrix via v_wmma_f32_16x16x32_f16 with
//      software-pipelined B-fragment loads + split fused top-5
//   3) conv/bn/relu stack, maxpool, antialiased bilinear resizes (fp32 direct)
// ---------------------------------------------------------------------------

typedef __attribute__((ext_vector_type(16))) _Float16 v16h;
typedef __attribute__((ext_vector_type(8)))  _Float16 v8h;
typedef __attribute__((ext_vector_type(8)))  float    v8f;

#define CORR_N    1369   // 37*37
#define CORR_NPAD 1376   // 86 * 16
#define CORR_C    384
#define CORR_KT   12     // 384 / 32
#define TOPK      5

// --------------------------- normalize -> f16 ------------------------------
// feats: (B, C, N) f32, channel stride N.  g: (B, NPAD, C) f16 row-major,
// rows >= N zero-filled.  One block per (b, n), 128 threads, C = 3*128.
__global__ __launch_bounds__(128)
void normalize_to_f16(const float* __restrict__ feats, _Float16* __restrict__ g,
                      int B, int C, int N, int NPAD)
{
    int n = blockIdx.x;
    int b = blockIdx.y;
    int t = threadIdx.x;
    _Float16* dst = g + ((size_t)b * NPAD + n) * C;
    if (n >= N) {
        for (int c = t; c < C; c += 128) dst[c] = (_Float16)0.0f;
        return;
    }
    const float* src = feats + (size_t)b * C * N + n;
    float v0 = src[(size_t)(t      ) * N];
    float v1 = src[(size_t)(t + 128) * N];
    float v2 = src[(size_t)(t + 256) * N];
    __shared__ float red[128];
    red[t] = v0 * v0 + v1 * v1 + v2 * v2;
    __syncthreads();
    for (int o = 64; o > 0; o >>= 1) {
        if (t < o) red[t] += red[t + o];
        __syncthreads();
    }
    float inv = 1.0f / (sqrtf(red[0]) + 1e-8f);
    dst[t]       = (_Float16)(v0 * inv);
    dst[t + 128] = (_Float16)(v1 * inv);
    dst[t + 256] = (_Float16)(v2 * inv);
}

// ----------------- self-correlation (WMMA) + fused top-5 -------------------
__device__ __forceinline__ void topk_insert(float (&t)[TOPK], float v)
{
    if (v > t[4]) {
        if (v > t[0])      { t[4]=t[3]; t[3]=t[2]; t[2]=t[1]; t[1]=t[0]; t[0]=v; }
        else if (v > t[1]) { t[4]=t[3]; t[3]=t[2]; t[2]=t[1]; t[1]=v; }
        else if (v > t[2]) { t[4]=t[3]; t[3]=t[2]; t[2]=v; }
        else if (v > t[3]) { t[4]=t[3]; t[3]=v; }
        else               { t[4]=v; }
    }
}

// B layout (32x16 f16): lane n<16 holds column N=n for K {0..7,16..23} (2 halves per
// VGPR); lane n+16 holds column N=n for K {8..15,24..31}.  Two b128 loads per K-tile.
__device__ __forceinline__ void load_bfrag(v16h (&buf)[CORR_KT],
                                           const _Float16* __restrict__ gb,
                                           int n0, int lrow, int hi)
{
    const _Float16* brow = gb + (size_t)(n0 + lrow) * CORR_C + hi * 8;
#pragma unroll
    for (int kt = 0; kt < CORR_KT; ++kt) {
        v8h b0 = *(const v8h*)(brow + kt * 32);
        v8h b1 = *(const v8h*)(brow + kt * 32 + 16);
#pragma unroll
        for (int i = 0; i < 8; ++i) { buf[kt][i] = b0[i]; buf[kt][i + 8] = b1[i]; }
    }
}

// One wave (32 lanes) per 16-row block per batch.
// A layout (16x32 f16, ISA 7.12.2): lane L<16 holds row M=L, K {0..7,16..23};
//   lane L+16 holds row M=L, K {8..15,24..31}; 2 halves per VGPR.
// C/D (16x16 f32): lane L holds column N=L&15, rows 8*(L>>4)+r in VGPR r.
__global__ __launch_bounds__(32)
void corr_topk_wmma(const _Float16* __restrict__ g, float* __restrict__ corr)
{
    const int b    = blockIdx.y;
    const int m0   = blockIdx.x * 16;
    const int lane = threadIdx.x;
    const int lrow = lane & 15;
    const int hi   = lane >> 4;

    const _Float16* gb = g + (size_t)b * CORR_NPAD * CORR_C;

    // Load all A fragments for this 16-row block (reused across all n-tiles).
    v16h afrag[CORR_KT];
    {
        const _Float16* arow = gb + (size_t)(m0 + lrow) * CORR_C;
#pragma unroll
        for (int kt = 0; kt < CORR_KT; ++kt) {
            const _Float16* p = arow + kt * 32;
            v8h lo = *(const v8h*)(p + hi * 8);        // K = hi*8 .. +7
            v8h up = *(const v8h*)(p + 16 + hi * 8);   // K = 16+hi*8 .. +7
            v16h a;
#pragma unroll
            for (int i = 0; i < 8; ++i) { a[i] = lo[i]; a[i + 8] = up[i]; }
            afrag[kt] = a;
        }
    }

    float tv[TOPK];
#pragma unroll
    for (int k = 0; k < TOPK; ++k) tv[k] = -3.0e38f;

    __shared__ float simLds[16][17];   // +1 pad against bank conflicts
    const int myrow = m0 + lrow;

    // Software pipeline: B fragments for tile n0 are loaded during the
    // previous tile's LDS scatter + top-k scan; WMMA chain then needs only
    // graduated waitcnts instead of a full stall per K-step.
    v16h bfrag[CORR_KT];
    load_bfrag(bfrag, gb, 0, lrow, hi);

    for (int n0 = 0; n0 < CORR_NPAD; n0 += 16) {
        v8f acc = {};
#pragma unroll
        for (int kt = 0; kt < CORR_KT; ++kt)
            acc = __builtin_amdgcn_wmma_f32_16x16x32_f16(
                false, afrag[kt], false, bfrag[kt], (short)0, acc, false, false);

        int n1 = n0 + 16;
        if (n1 < CORR_NPAD) {
            load_bfrag(bfrag, gb, n1, lrow, hi);       // prefetch next tile (regs)
            if (n1 + 16 < CORR_NPAD)                   // pull tile-after-next to cache
                __builtin_prefetch((const void*)(gb + (size_t)(n1 + 16 + lrow) * CORR_C
                                                 + hi * 192), 0, 1);
        }

        // Scatter D tile to LDS as sim[m][n] (m,n local to tile).
#pragma unroll
        for (int r = 0; r < 8; ++r) simLds[hi * 8 + r][lrow] = acc[r];
        __syncthreads();

        // Split top-k scan: half-wave hi handles columns hi*8 .. hi*8+7.
        if (myrow < CORR_N) {
#pragma unroll
            for (int j = 0; j < 8; ++j) {
                int   n = n0 + hi * 8 + j;
                float v = simLds[lrow][hi * 8 + j];
                if (n < CORR_N && n != myrow) topk_insert(tv, v);
            }
        }
        __syncthreads();
    }

    // Merge hi=1 partial top-5 into hi=0 lane of the same row via LDS.
    if (hi == 1) {
#pragma unroll
        for (int k = 0; k < TOPK; ++k) simLds[lrow][k] = tv[k];
    }
    __syncthreads();
    if (hi == 0 && myrow < CORR_N) {
#pragma unroll
        for (int k = 0; k < TOPK; ++k) topk_insert(tv, simLds[lrow][k]);
        float* o = corr + (size_t)b * TOPK * CORR_N + myrow;
        o[0 * CORR_N] = tv[0];
        o[1 * CORR_N] = tv[1];
        o[2 * CORR_N] = tv[2];
        o[3 * CORR_N] = tv[3];
        o[4 * CORR_N] = tv[4];
    }
}

// --------------------- direct conv + BN + ReLU (fp32) ----------------------
// Optional second input (channel concat).  bn_g==nullptr -> plain conv+bias.
__global__ void conv_bn_relu(const float* __restrict__ in1, int Cin1,
                             const float* __restrict__ in2, int Cin2,
                             const float* __restrict__ w, const float* __restrict__ bias,
                             const float* __restrict__ bn_g, const float* __restrict__ bn_be,
                             const float* __restrict__ bn_m, const float* __restrict__ bn_v,
                             float* __restrict__ out,
                             int B, int Cout, int H, int W, int ksize)
{
    int idx = blockIdx.x * blockDim.x + threadIdx.x;
    int HW = H * W;
    int total = B * Cout * HW;
    if (idx >= total) return;
    int x  = idx % W;
    int y  = (idx / W) % H;
    int oc = (idx / HW) % Cout;
    int b  = idx / (HW * Cout);
    int Cin = Cin1 + Cin2;
    int pad = ksize >> 1;

    float acc = 0.0f;
    for (int ic = 0; ic < Cin; ++ic) {
        const float* src = (ic < Cin1)
            ? (in1 + ((size_t)b * Cin1 + ic) * HW)
            : (in2 + ((size_t)b * Cin2 + (ic - Cin1)) * HW);
        const float* wk = w + ((size_t)oc * Cin + ic) * ksize * ksize;
        for (int dy = 0; dy < ksize; ++dy) {
            int yy = y + dy - pad;
            if (yy < 0 || yy >= H) continue;
            for (int dx = 0; dx < ksize; ++dx) {
                int xx = x + dx - pad;
                if (xx < 0 || xx >= W) continue;
                acc = fmaf(src[yy * W + xx], wk[dy * ksize + dx], acc);
            }
        }
    }
    acc += bias[oc];
    if (bn_g) {
        float scale = bn_g[oc] * rsqrtf(bn_v[oc] + 1e-5f);
        float shift = bn_be[oc] - bn_m[oc] * scale;
        acc = fmaxf(acc * scale + shift, 0.0f);
    }
    out[idx] = acc;
}

// ------------------------------ 2x2 max pool -------------------------------
__global__ void maxpool2x2(const float* __restrict__ in, float* __restrict__ out,
                           int B, int C, int H, int W)
{
    int Ho = H / 2, Wo = W / 2;
    int idx = blockIdx.x * blockDim.x + threadIdx.x;
    int total = B * C * Ho * Wo;
    if (idx >= total) return;
    int x  = idx % Wo;
    int y  = (idx / Wo) % Ho;
    int bc = idx / (Wo * Ho);
    const float* s = in + (size_t)bc * H * W + (2 * y) * W + 2 * x;
    float m = fmaxf(fmaxf(s[0], s[1]), fmaxf(s[W], s[W + 1]));
    out[idx] = m;
}

// ---------------- bilinear resize (jax.image.resize semantics) -------------
// Half-pixel centers; antialias triangle filter on downscale (kernel scale
// s = min(out/in, 1)); per-pixel weight renormalization over in-range taps.
__global__ void resize_bilinear_aa(const float* __restrict__ in, float* __restrict__ out,
                                   int B, int C, int Hi, int Wi, int Ho, int Wo)
{
    int idx = blockIdx.x * blockDim.x + threadIdx.x;
    int total = B * C * Ho * Wo;
    if (idx >= total) return;
    int x  = idx % Wo;
    int y  = (idx / Wo) % Ho;
    int bc = idx / (Wo * Ho);
    const float* src = in + (size_t)bc * Hi * Wi;

    float sy = (float)Ho / (float)Hi; if (sy > 1.0f) sy = 1.0f;
    float sx = (float)Wo / (float)Wi; if (sx > 1.0f) sx = 1.0f;
    float cy = (y + 0.5f) * (float)Hi / (float)Ho - 0.5f;
    float cx = (x + 0.5f) * (float)Wi / (float)Wo - 0.5f;
    float ry = 1.0f / sy, rx = 1.0f / sx;

    int y0 = (int)ceilf(cy - ry);  if (y0 < 0) y0 = 0;
    int y1 = (int)floorf(cy + ry); if (y1 > Hi - 1) y1 = Hi - 1;
    int x0 = (int)ceilf(cx - rx);  if (x0 < 0) x0 = 0;
    int x1 = (int)floorf(cx + rx); if (x1 > Wi - 1) x1 = Wi - 1;

    float acc = 0.0f, wacc = 0.0f;
    for (int yy = y0; yy <= y1; ++yy) {
        float wy = 1.0f - fabsf((float)yy - cy) * sy;
        if (wy <= 0.0f) continue;
        for (int xx = x0; xx <= x1; ++xx) {
            float wx = 1.0f - fabsf((float)xx - cx) * sx;
            if (wx <= 0.0f) continue;
            float wgt = wy * wx;
            acc  += wgt * src[yy * Wi + xx];
            wacc += wgt;
        }
    }
    out[idx] = acc / wacc;
}

// ---------------------------------------------------------------------------
extern "C" void kernel_launch(void* const* d_in, const int* in_sizes, int n_in,
                              void* d_out, int out_size, void* d_ws, size_t ws_size,
                              hipStream_t stream)
{
    (void)in_sizes; (void)n_in; (void)out_size; (void)ws_size;

    const int B = 16, C = 384, H = 37, W = 37;
    const int N = H * W;           // 1369
    const int NPAD = CORR_NPAD;    // 1376
    const int Hp = 18, Wp = 18, Np = Hp * Wp;

    // ---- inputs: JAX pytree flatten order (dict keys sorted, recursive) ----
    const float* feats = (const float*)d_in[0];
#define CONV_BN(name, i)                                   \
    const float* name##_b  = (const float*)d_in[(i)];      \
    const float* name##_be = (const float*)d_in[(i) + 1];  \
    const float* name##_g  = (const float*)d_in[(i) + 2];  \
    const float* name##_m  = (const float*)d_in[(i) + 3];  \
    const float* name##_v  = (const float*)d_in[(i) + 4];  \
    const float* name##_w  = (const float*)d_in[(i) + 5];
    CONV_BN(bott1, 1)    // bott.c1
    CONV_BN(bott2, 7)    // bott.c2
    CONV_BN(ch1, 13)
    CONV_BN(ch2, 19)
    const float* ch3_b = (const float*)d_in[25];
    const float* ch3_w = (const float*)d_in[26];
    CONV_BN(dec1, 27)
    CONV_BN(dec2, 33)
    CONV_BN(enc1c1, 39)
    CONV_BN(enc1c2, 45)
    CONV_BN(enc2c1, 51)
    CONV_BN(enc2c2, 57)
    const float* outc_b = (const float*)d_in[63];
    const float* outc_w = (const float*)d_in[64];
#undef CONV_BN

    // ---- workspace bump allocator ----
    char* ws = (char*)d_ws;
    size_t off = 0;
    auto alloc = [&](size_t bytes) -> void* {
        void* p = ws + off;
        off += (bytes + 255) & ~(size_t)255;
        return p;
    };
    _Float16* g = (_Float16*)alloc((size_t)B * NPAD * C * sizeof(_Float16));
    float* corr = (float*)alloc((size_t)B * TOPK * N * 4);
    float* t1   = (float*)alloc((size_t)B * 128 * N * 4);
    float* t2   = (float*)alloc((size_t)B * 128 * N * 4);
    float* sal  = (float*)alloc((size_t)B * N * 4);
    float* e1t  = (float*)alloc((size_t)B * 64 * N * 4);
    float* e1   = (float*)alloc((size_t)B * 64 * N * 4);
    float* p1   = (float*)alloc((size_t)B * 64 * Np * 4);
    float* e2t  = (float*)alloc((size_t)B * 128 * Np * 4);
    float* e2   = (float*)alloc((size_t)B * 128 * Np * 4);
    float* bb   = (float*)alloc((size_t)B * 384 * Np * 4);
    float* bt1  = (float*)alloc((size_t)B * 128 * Np * 4);
    float* btt  = (float*)alloc((size_t)B * 128 * Np * 4);
    float* r1   = (float*)alloc((size_t)B * 128 * 36 * 36 * 4);
    float* d1   = (float*)alloc((size_t)B * 64 * 36 * 36 * 4);
    float* r2   = (float*)alloc((size_t)B * 64 * 72 * 72 * 4);
    float* d2   = (float*)alloc((size_t)B * 32 * 72 * 72 * 4);
    float* o72  = (float*)alloc((size_t)B * 72 * 72 * 4);

    auto conv = [&](const float* i1, int c1, const float* i2, int c2,
                    const float* wp, const float* bp,
                    const float* gp, const float* bep, const float* mp, const float* vp,
                    float* op, int co, int hh, int wwid, int k) {
        int total = B * co * hh * wwid;
        conv_bn_relu<<<dim3((total + 255) / 256), 256, 0, stream>>>(
            i1, c1, i2, c2, wp, bp, gp, bep, mp, vp, op, B, co, hh, wwid, k);
    };
    auto resize = [&](const float* ip, float* op, int cc, int hi, int wi, int ho, int wo) {
        int total = B * cc * ho * wo;
        resize_bilinear_aa<<<dim3((total + 255) / 256), 256, 0, stream>>>(
            ip, op, B, cc, hi, wi, ho, wo);
    };

    // 1) normalize + f16 pack
    normalize_to_f16<<<dim3(NPAD, B), 128, 0, stream>>>(feats, g, B, C, N, NPAD);
    // 2) WMMA self-correlation + top-5
    corr_topk_wmma<<<dim3(NPAD / 16, B), 32, 0, stream>>>(g, corr);
    // 3) CorrHead
    conv(corr, TOPK, nullptr, 0, ch1_w, ch1_b, ch1_g, ch1_be, ch1_m, ch1_v, t1, 128, H, W, 3);
    conv(t1, 128, nullptr, 0, ch2_w, ch2_b, ch2_g, ch2_be, ch2_m, ch2_v, t2, 128, H, W, 3);
    conv(t2, 128, nullptr, 0, ch3_w, ch3_b, nullptr, nullptr, nullptr, nullptr, sal, 1, H, W, 1);
    // 4) TinyDecoder
    conv(sal, 1, nullptr, 0, enc1c1_w, enc1c1_b, enc1c1_g, enc1c1_be, enc1c1_m, enc1c1_v, e1t, 64, H, W, 3);
    conv(e1t, 64, nullptr, 0, enc1c2_w, enc1c2_b, enc1c2_g, enc1c2_be, enc1c2_m, enc1c2_v, e1, 64, H, W, 3);
    {
        int total = B * 64 * Hp * Wp;
        maxpool2x2<<<dim3((total + 255) / 256), 256, 0, stream>>>(e1, p1, B, 64, H, W);
    }
    conv(p1, 64, nullptr, 0, enc2c1_w, enc2c1_b, enc2c1_g, enc2c1_be, enc2c1_m, enc2c1_v, e2t, 128, Hp, Wp, 3);
    conv(e2t, 128, nullptr, 0, enc2c2_w, enc2c2_b, enc2c2_g, enc2c2_be, enc2c2_m, enc2c2_v, e2, 128, Hp, Wp, 3);
    resize(feats, bb, 384, H, W, Hp, Wp);                           // backbone skip (antialiased downscale)
    conv(e2, 128, bb, 384, bott1_w, bott1_b, bott1_g, bott1_be, bott1_m, bott1_v, bt1, 128, Hp, Wp, 3);
    conv(bt1, 128, nullptr, 0, bott2_w, bott2_b, bott2_g, bott2_be, bott2_m, bott2_v, btt, 128, Hp, Wp, 3);
    resize(btt, r1, 128, Hp, Wp, 36, 36);
    conv(r1, 128, nullptr, 0, dec1_w, dec1_b, dec1_g, dec1_be, dec1_m, dec1_v, d1, 64, 36, 36, 3);
    resize(d1, r2, 64, 36, 36, 72, 72);
    conv(r2, 64, nullptr, 0, dec2_w, dec2_b, dec2_g, dec2_be, dec2_m, dec2_v, d2, 32, 72, 72, 3);
    conv(d2, 32, nullptr, 0, outc_w, outc_b, nullptr, nullptr, nullptr, nullptr, o72, 1, 72, 72, 1);
    // 5) final upsample to 518x518
    resize(o72, (float*)d_out, 1, 72, 72, 14 * H, 14 * W);
}
